// SASelfAttention_45183055954033
// MI455X (gfx1250) — compile-verified, matrix operand order checked by reading, not run on hardware
//
#include <hip/hip_runtime.h>
#include <cstdint>

typedef __attribute__((ext_vector_type(16))) _Float16 v16h;
typedef __attribute__((ext_vector_type(8)))  _Float16 v8h;
typedef __attribute__((ext_vector_type(8)))  float    v8f;
typedef unsigned int u32x4 __attribute__((ext_vector_type(4)));
typedef int          i32x8 __attribute__((ext_vector_type(8)));
typedef int          i32x4 __attribute__((ext_vector_type(4)));

#define B_   4
#define C_   128
#define H_   32
#define W_   256
#define S_   (H_ * W_)   // 8192
#define T_   (S_ / 2)    // 4096
#define CQK_ 16
#define LOG2E_ 1.44269504088896340736f

#if defined(__has_builtin)
#if __has_builtin(__builtin_amdgcn_tensor_load_to_lds)
#define HAVE_TDM 1
#endif
#endif

// ---------------------------------------------------------------------------
// helpers
// ---------------------------------------------------------------------------
__device__ __forceinline__ v16h pack16(v8h lo, v8h hi) {
  v16h r;
#pragma unroll
  for (int i = 0; i < 8; ++i) { r[i] = lo[i]; r[i + 8] = hi[i]; }
  return r;
}
__device__ __forceinline__ v16h pack16z(v8h lo) {
  v16h r;
#pragma unroll
  for (int i = 0; i < 8; ++i) { r[i] = lo[i]; r[i + 8] = (_Float16)0.0f; }
  return r;
}

// 16-lane all-reduce max via fused DPP max (1 VALU per butterfly step).
__device__ __forceinline__ float dpp_max16(float v) {
  float t;
  asm volatile("v_max_num_f32_dpp %0, %1, %1 quad_perm:[1,0,3,2] row_mask:0xf bank_mask:0xf bound_ctrl:1"
               : "=v"(t) : "v"(v));
  asm volatile("v_max_num_f32_dpp %0, %1, %1 quad_perm:[2,3,0,1] row_mask:0xf bank_mask:0xf bound_ctrl:1"
               : "=v"(v) : "v"(t));
  asm volatile("v_max_num_f32_dpp %0, %1, %1 row_half_mirror row_mask:0xf bank_mask:0xf bound_ctrl:1"
               : "=v"(t) : "v"(v));
  asm volatile("v_max_num_f32_dpp %0, %1, %1 row_mirror row_mask:0xf bank_mask:0xf bound_ctrl:1"
               : "=v"(v) : "v"(t));
  return v;
}

#if HAVE_TDM
// TDM: DMA a 2-D tile (rows of tile_d0 contiguous elems, tile_d1 rows with
// row stride d0_stride elems) of 2-byte elements into LDS.
// D# per CDNA5 ISA 8.3/8.4: group0 = {flags, lds_addr, gaddr_lo, gaddr_hi|type},
// group1 = {data_size, dims/tiles/strides}, remaining groups = 0 (<=2-D tensor).
__device__ __forceinline__ void tdm_load_2d(uint32_t lds_addr, const _Float16* gp,
                                            uint32_t tensor_d0, uint32_t tensor_d1,
                                            uint32_t tile_d0, uint32_t tile_d1,
                                            uint32_t d0_stride) {
  uint64_t ga = (uint64_t)(uintptr_t)gp;
  u32x4 g0;
  g0[0] = 1u;                                                   // count=1
  g0[1] = lds_addr;                                             // LDS byte addr
  g0[2] = (uint32_t)ga;                                         // gaddr[31:0]
  g0[3] = ((uint32_t)(ga >> 32) & 0x01FFFFFFu) | 0x80000000u;   // gaddr[56:32]|type=2
  i32x8 g1;
  g1[0] = 0x00010000;                                           // data_size=1 (2B)
  g1[1] = (int)((tensor_d0 & 0xFFFFu) << 16);                   // tensor_dim0 lo
  g1[2] = (int)((tensor_d0 >> 16) | ((tensor_d1 & 0xFFFFu) << 16));
  g1[3] = (int)((tensor_d1 >> 16) | (tile_d0 << 16));           // tile_dim0
  g1[4] = (int)tile_d1;                                         // tile_dim1 (dim2=0)
  g1[5] = (int)d0_stride;                                       // dim0 stride lo
  g1[6] = 0;
  g1[7] = 0;
  i32x4 z4 = {0, 0, 0, 0};
  i32x8 z8 = {0, 0, 0, 0, 0, 0, 0, 0};
  __builtin_amdgcn_tensor_load_to_lds(g0, g1, z4, z4, z8, 0);
}
#endif

// ---------------------------------------------------------------------------
// Projection kernels (1x1 convs, f32 -> f16, with (1,2) max-pool for phi/g)
// theta: [B,S,16] (pre-scaled by log2(e) -> softmax runs in exp2 domain)
// phi  : [B,T,16]  g: [B,C,T]
// ---------------------------------------------------------------------------
__global__ void k_proj_theta(const float* __restrict__ x, const float* __restrict__ w,
                             _Float16* __restrict__ th) {
  int idx = blockIdx.x * blockDim.x + threadIdx.x;
  if (idx >= B_ * CQK_ * S_) return;
  int s = idx % S_;
  int o = (idx / S_) % CQK_;
  int b = idx / (S_ * CQK_);
  const float* xc = x + (size_t)b * C_ * S_ + s;
  const float* wr = w + o * C_;
  float acc = 0.0f;
#pragma unroll 8
  for (int c = 0; c < C_; ++c) acc += wr[c] * xc[(size_t)c * S_];
  th[((size_t)(b * S_ + s)) * CQK_ + o] = (_Float16)(acc * LOG2E_);
}

__global__ void k_proj_phi(const float* __restrict__ x, const float* __restrict__ w,
                           _Float16* __restrict__ ph) {
  int idx = blockIdx.x * blockDim.x + threadIdx.x;
  if (idx >= B_ * CQK_ * T_) return;
  int t = idx % T_;
  int o = (idx / T_) % CQK_;
  int b = idx / (T_ * CQK_);
  int h = t >> 7, w2 = t & 127;
  int s0 = h * W_ + 2 * w2;
  const float* xc = x + (size_t)b * C_ * S_ + s0;
  const float* wr = w + o * C_;
  float a0 = 0.0f, a1 = 0.0f;
#pragma unroll 8
  for (int c = 0; c < C_; ++c) {
    float wv = wr[c];
    a0 += wv * xc[(size_t)c * S_];
    a1 += wv * xc[(size_t)c * S_ + 1];
  }
  ph[((size_t)(b * T_ + t)) * CQK_ + o] = (_Float16)fmaxf(a0, a1);
}

__global__ void k_proj_g(const float* __restrict__ x, const float* __restrict__ w,
                         _Float16* __restrict__ g) {
  int idx = blockIdx.x * blockDim.x + threadIdx.x;
  if (idx >= B_ * C_ * T_) return;
  int t = idx % T_;
  int co = (idx / T_) % C_;
  int b = idx / (T_ * C_);
  int h = t >> 7, w2 = t & 127;
  int s0 = h * W_ + 2 * w2;
  const float* xc = x + (size_t)b * C_ * S_ + s0;
  const float* wr = w + co * C_;
  float a0 = 0.0f, a1 = 0.0f;
#pragma unroll 8
  for (int c = 0; c < C_; ++c) {
    float wv = wr[c];
    a0 += wv * xc[(size_t)c * S_];
    a1 += wv * xc[(size_t)c * S_ + 1];
  }
  g[((size_t)(b * C_ + co)) * T_ + t] = (_Float16)fmaxf(a0, a1);
}

__global__ void k_cvt_h16(const float* __restrict__ in, _Float16* __restrict__ o, int n) {
  int i = blockIdx.x * blockDim.x + threadIdx.x;
  if (i < n) o[i] = (_Float16)in[i];
}

// ---------------------------------------------------------------------------
// Fused flash attention, block-cooperative, double-buffered K/V in LDS:
//  8 waves/block on 8 query tiles of one batch. Per 32-key step the NEXT
//  phi (1KB) + g (8KB) tiles are staged into LDS by the Tensor Data Mover
//  (wave 0 issues TENSOR_LOAD_TO_LDS; TENSORcnt + barrier publish), fully
//  overlapped with this step's 2 QK WMMAs + DPP-max/exp2 softmax +
//  1 rowsum WMMA + 8 PV WMMAs.
// ---------------------------------------------------------------------------
__global__ __launch_bounds__(256) void k_attn(const _Float16* __restrict__ th,
                                              const _Float16* __restrict__ ph,
                                              const _Float16* __restrict__ g,
                                              _Float16* __restrict__ ob) {
  __shared__ _Float16 Pl[8][16 * 32];     // per-wave 16x32 P tiles  (8 KB)
  __shared__ _Float16 Gt[2][C_][32];      // g tiles, [c][t]        (16 KB)
  __shared__ _Float16 Pt[2][32][CQK_];    // phi tiles, [t][ch]      (2 KB)

  const int tid  = threadIdx.x;
  const int lane = tid & 31;
  const int wv   = tid >> 5;
  const int qt   = blockIdx.x * 8 + wv;          // 2048 query tiles total
  const int b    = qt / (S_ / 16);               // uniform per block
  const int sb   = (qt % (S_ / 16)) * 16;
  const int half = lane >> 4;
  const int l16  = lane & 15;

  const _Float16* gb  = g + (size_t)b * C_ * T_;
  const _Float16* phb = ph + (size_t)b * T_ * CQK_;

  // stage key/value tiles for keys [t, t+32) into LDS buffer `buf`
  auto stage = [&](int buf, int t) {
#if HAVE_TDM
    if (wv == 0) {
      tdm_load_2d((uint32_t)(uintptr_t)&Gt[buf][0][0], gb + t,
                  T_, C_, 32, C_, T_);
      tdm_load_2d((uint32_t)(uintptr_t)&Pt[buf][0][0], phb + (size_t)t * CQK_,
                  CQK_, T_, CQK_, 32, CQK_);
    }
#else
    int c0 = tid >> 2, o0 = (tid & 3) * 8;
    *(v8h*)&Gt[buf][c0][o0] = *(const v8h*)(gb + (size_t)c0 * T_ + t + o0);
    int j1 = tid + 256;
    int c1 = j1 >> 2, o1 = (j1 & 3) * 8;
    *(v8h*)&Gt[buf][c1][o1] = *(const v8h*)(gb + (size_t)c1 * T_ + t + o1);
    if (tid < 64) {
      int kt = tid >> 1, co = (tid & 1) * 8;
      *(v8h*)&Pt[buf][kt][co] = *(const v8h*)(phb + (size_t)(t + kt) * CQK_ + co);
    }
#endif
  };
  auto stage_drain = [&]() {
#if HAVE_TDM
    if (wv == 0) __builtin_amdgcn_s_wait_tensorcnt(0);
#endif
    __syncthreads();
  };

  // A operand: theta rows (M = query), K = channel 0..15, padded to 32.
  v16h A;
  {
    const _Float16* p = th + ((size_t)(b * S_ + sb + l16)) * CQK_ + half * 8;
    A = pack16z(*(const v8h*)p);
  }

  v16h Ones;
#pragma unroll
  for (int i = 0; i < 16; ++i) Ones[i] = (_Float16)1.0f;

  float m[8];
  v8f O[8], L;
#pragma unroll
  for (int r = 0; r < 8; ++r) { m[r] = -3.0e38f; L[r] = 0.0f; }
#pragma unroll
  for (int n = 0; n < 8; ++n)
#pragma unroll
    for (int r = 0; r < 8; ++r) O[n][r] = 0.0f;

  v8f zc = {};

  stage(0, 0);
  stage_drain();

  for (int it = 0; it < T_ / 32; ++it) {
    const int t   = it * 32;
    const int cur = it & 1, nxt = cur ^ 1;
    if (t + 32 < T_) stage(nxt, t + 32);  // overlap DMA of next tile

    // ---- scores: two 16x16 tiles over keys t..t+31 (B from LDS) -----------
    v8f S0, S1;
    {
      v16h B0 = pack16z(*(const v8h*)&Pt[cur][l16][half * 8]);
      v16h B1 = pack16z(*(const v8h*)&Pt[cur][16 + l16][half * 8]);
      S0 = __builtin_amdgcn_wmma_f32_16x16x32_f16(false, A, false, B0, (short)0, zc, false, false);
      S1 = __builtin_amdgcn_wmma_f32_16x16x32_f16(false, A, false, B1, (short)0, zc, false, false);
    }

    // ---- online softmax update (exp2 domain, fused DPP max) ---------------
#pragma unroll
    for (int r = 0; r < 8; ++r) {
      float v  = dpp_max16(fmaxf(S0[r], S1[r]));
      float mn = fmaxf(m[r], v);
      float sc = __builtin_amdgcn_exp2f(m[r] - mn);
      m[r] = mn;
      float p0 = __builtin_amdgcn_exp2f(S0[r] - mn);
      float p1 = __builtin_amdgcn_exp2f(S1[r] - mn);
      L[r] *= sc;
#pragma unroll
      for (int n = 0; n < 8; ++n) O[n][r] *= sc;
      int row = r + half * 8;
      Pl[wv][row * 32 + l16]      = (_Float16)p0;
      Pl[wv][row * 32 + 16 + l16] = (_Float16)p1;
    }
    asm volatile("s_wait_dscnt 0" ::: "memory");

    // ---- read P back in A-operand layout (M=query, K=key offset) ----------
    v16h Pa;
    {
      const _Float16* pr = &Pl[wv][l16 * 32 + half * 8];
      Pa = pack16(*(const v8h*)pr, *(const v8h*)(pr + 16));
    }

    // ---- denominator: L += rowsum(P) via WMMA with all-ones B -------------
    L = __builtin_amdgcn_wmma_f32_16x16x32_f16(false, Pa, false, Ones, (short)0, L, false, false);

    // ---- O += P * g^T across 8 channel tiles (B from LDS) -----------------
#pragma unroll
    for (int n = 0; n < 8; ++n) {
      const _Float16* gp = &Gt[cur][n * 16 + l16][half * 8];
      v16h Gv = pack16(*(const v8h*)gp, *(const v8h*)(gp + 16));
      O[n] = __builtin_amdgcn_wmma_f32_16x16x32_f16(false, Pa, false, Gv, (short)0, O[n], false, false);
    }

    stage_drain();  // publish next tile; all waves done with `cur`
  }

  // ---- normalize and store o as [B,S,128] f16 ------------------------------
  float inv[8];
#pragma unroll
  for (int r = 0; r < 8; ++r) inv[r] = 1.0f / L[r];
#pragma unroll
  for (int n = 0; n < 8; ++n)
#pragma unroll
    for (int r = 0; r < 8; ++r) {
      int s = sb + r + half * 8;
      ob[((size_t)(b * S_ + s)) * C_ + n * 16 + l16] = (_Float16)(O[n][r] * inv[r]);
    }
}

// ---------------------------------------------------------------------------
// Output projection: out = gamma * (w_o @ o) + x, one wave per 16-pixel tile.
// ---------------------------------------------------------------------------
__global__ __launch_bounds__(256) void k_out(const _Float16* __restrict__ ob,
                                             const _Float16* __restrict__ woh,
                                             const float* __restrict__ x,
                                             const float* __restrict__ gscal,
                                             float* __restrict__ out) {
  const int lane = threadIdx.x & 31;
  const int wv   = threadIdx.x >> 5;
  const int pt   = blockIdx.x * 8 + wv;          // 2048 pixel tiles total
  const int b    = pt / (S_ / 16);
  const int pb   = (pt % (S_ / 16)) * 16;
  const int half = lane >> 4;
  const int l16  = lane & 15;
  const float gamma = *gscal;

  v8f Acc[8];
#pragma unroll
  for (int n = 0; n < 8; ++n)
#pragma unroll
    for (int r = 0; r < 8; ++r) Acc[n][r] = 0.0f;

#pragma unroll
  for (int kc = 0; kc < 4; ++kc) {
    const _Float16* op = ob + ((size_t)(b * S_ + pb + l16)) * C_ + kc * 32 + half * 8;
    v16h Bv = pack16(*(const v8h*)op, *(const v8h*)(op + 16));
#pragma unroll
    for (int mt = 0; mt < 8; ++mt) {
      const _Float16* wp = woh + (size_t)(mt * 16 + l16) * C_ + kc * 32 + half * 8;
      v16h Av = pack16(*(const v8h*)wp, *(const v8h*)(wp + 16));
      Acc[mt] = __builtin_amdgcn_wmma_f32_16x16x32_f16(false, Av, false, Bv, (short)0, Acc[mt], false, false);
    }
  }

#pragma unroll
  for (int mt = 0; mt < 8; ++mt)
#pragma unroll
    for (int r = 0; r < 8; ++r) {
      int oc = mt * 16 + r + half * 8;
      int p  = pb + l16;
      size_t xi = ((size_t)(b * C_ + oc)) * S_ + p;
      out[xi] = gamma * Acc[mt][r] + x[xi];
    }
}

// ---------------------------------------------------------------------------
extern "C" void kernel_launch(void* const* d_in, const int* in_sizes, int n_in,
                              void* d_out, int out_size, void* d_ws, size_t ws_size,
                              hipStream_t stream) {
  const float* x       = (const float*)d_in[0];
  const float* w_theta = (const float*)d_in[1];
  const float* w_phi   = (const float*)d_in[2];
  const float* w_g     = (const float*)d_in[3];
  const float* w_o     = (const float*)d_in[4];
  const float* gamma   = (const float*)d_in[5];
  float* out = (float*)d_out;

  char* ws = (char*)d_ws;
  size_t off = 0;
  auto carve = [&](size_t bytes) -> char* {
    char* p = ws + off;
    off += (bytes + 255) & ~(size_t)255;
    return p;
  };
  _Float16* th  = (_Float16*)carve((size_t)B_ * S_ * CQK_ * 2);  // 1 MB
  _Float16* ph  = (_Float16*)carve((size_t)B_ * T_ * CQK_ * 2);  // 0.5 MB
  _Float16* g   = (_Float16*)carve((size_t)B_ * C_ * T_ * 2);    // 4 MB
  _Float16* ob  = (_Float16*)carve((size_t)B_ * S_ * C_ * 2);    // 8 MB
  _Float16* woh = (_Float16*)carve((size_t)C_ * C_ * 2);         // 32 KB
  (void)ws_size; (void)in_sizes; (void)n_in; (void)out_size;

  // projections
  k_proj_theta<<<(B_ * CQK_ * S_ + 255) / 256, 256, 0, stream>>>(x, w_theta, th);
  k_proj_phi  <<<(B_ * CQK_ * T_ + 255) / 256, 256, 0, stream>>>(x, w_phi, ph);
  k_proj_g    <<<(B_ * C_  * T_ + 255) / 256, 256, 0, stream>>>(x, w_g, g);
  k_cvt_h16   <<<(C_ * C_ + 255) / 256, 256, 0, stream>>>(w_o, woh, C_ * C_);

  // fused flash attention: 2048 query tiles, 8 waves/block,
  // double-buffered K/V tiles staged by the Tensor Data Mover
  k_attn<<<(B_ * (S_ / 16)) / 8, 256, 0, stream>>>(th, ph, g, ob);

  // output projection + residual: 2048 pixel tiles, 8 waves/block
  k_out<<<(B_ * (S_ / 16)) / 8, 256, 0, stream>>>(ob, woh, x, gamma, out);
}
